// Attention_81449759801973
// MI455X (gfx1250) — matrix operand order repaired and checked by reading, not compiled, not run on hardware
//
#include <hip/hip_runtime.h>
#include <hip/hip_bf16.h>

// ---------------------------------------------------------------------------
// Types for CDNA5 WMMA (wave32, v_wmma_f32_16x16x32_bf16)
// ---------------------------------------------------------------------------
typedef __bf16 bf16;
typedef __attribute__((ext_vector_type(16))) __bf16 bf16x16;
typedef __attribute__((ext_vector_type(8)))  float  f32x8;
typedef __attribute__((ext_vector_type(4)))  float  f32x4;
typedef __attribute__((ext_vector_type(4)))  unsigned int u32x4;
typedef __attribute__((ext_vector_type(4)))  int i32x4;
typedef __attribute__((ext_vector_type(8)))  int i32x8;

union Frag {
    bf16x16 v;
    u32x4   u[2];
};

__device__ inline f32x8 wmma_bf16(const Frag& a, const Frag& b, f32x8 c) {
    return __builtin_amdgcn_wmma_f32_16x16x32_bf16(
        /*neg_a=*/false, a.v, /*neg_b=*/false, b.v,
        /*c_mod=*/(short)0, c, /*reuse_a=*/false, /*reuse_b=*/false);
}

// A fragment (16x32 bf16), source row-major, row stride `rs` elements.
// ISA layout: lane l<16 -> M=l, K = {0..7,16..23}; lane l+16 -> K={8..15,24..31}
__device__ inline void load_a_frag(Frag& f, const bf16* base, int rs, int l, int hi) {
    const bf16* row = base + (long long)l * rs;
    f.u[0] = *(const u32x4*)(row + hi * 8);
    f.u[1] = *(const u32x4*)(row + 16 + hi * 8);
}

// B fragment (32x16 bf16), source stored K-contiguous per N-row ("BT" layout),
// row stride `rs`. ISA layout: lane -> N=l, K = {8hi..8hi+7, 16+8hi..16+8hi+7}.
__device__ inline void load_b_frag(Frag& f, const bf16* base, int rs, int l, int hi) {
    const bf16* row = base + (long long)l * rs;
    f.u[0] = *(const u32x4*)(row + hi * 8);
    f.u[1] = *(const u32x4*)(row + 16 + hi * 8);
}

// ---------------------------------------------------------------------------
// Tensor Data Mover: async 2D bf16 tile load (tile_d0 x tile_d1, row stride
// `stride0` elements) from global into LDS. D# per CDNA5 ISA §8.
// 6-arg builtin on this toolchain:
//   (uint32x4 g0, int32x8 g1, int32x4 g2, int32x4 g3, int32x8 g4, int cpol)
// ---------------------------------------------------------------------------
#if defined(__has_builtin)
#  if __has_builtin(__builtin_amdgcn_tensor_load_to_lds) && \
      __has_builtin(__builtin_amdgcn_s_wait_tensorcnt)
#    define USE_TDM 1
#  endif
#endif

#ifdef USE_TDM
__device__ inline void tdm_load_2d_bf16(unsigned lds_byte_addr, const bf16* gptr,
                                        unsigned tile_d0, unsigned tile_d1,
                                        unsigned stride0) {
    unsigned long long ga = (unsigned long long)(size_t)gptr;
    // Group 0: count=1 | lds_addr | global_addr[56:0] | type=2 ("image")
    u32x4 g0 = { 1u,
                 lds_byte_addr,
                 (unsigned)(ga & 0xFFFFFFFFu),
                 (unsigned)((ga >> 32) & 0x01FFFFFFu) | (2u << 30) };
    // Group 1: wg_mask=0, data_size=1 (2B), no barrier/iterate/pad;
    // tensor dims = tile dims (tile in-bounds by construction).
    int w1 = (int)((tile_d0 & 0xFFFFu) << 16);               // td0[15:0] in [63:48]
    int w2 = (int)(((tile_d0 >> 16) & 0xFFFFu) |
                   ((tile_d1 & 0xFFFFu) << 16));             // td0 hi | td1 lo
    int w3 = (int)(((tile_d1 >> 16) & 0xFFFFu) |
                   ((tile_d0 & 0xFFFFu) << 16));             // td1 hi | tile_dim0
    int w4 = (int)(tile_d1 & 0xFFFFu);                       // tile_dim1 (tile_dim2=0)
    i32x8 g1 = { 1 << 16, w1, w2, w3, w4, (int)stride0, 0, 0 };
    i32x4 gz4 = { 0, 0, 0, 0 };                              // 2D: groups 2/3 unused
    i32x8 gz8 = { 0, 0, 0, 0, 0, 0, 0, 0 };
    __builtin_amdgcn_tensor_load_to_lds(g0, g1, gz4, gz4, gz8, 0);
}
#endif

// ---------------------------------------------------------------------------
// Weight conversion + transpose: y[n][k] = (bf16)x[k][n], 512x512
// ---------------------------------------------------------------------------
__global__ void cvt_t_bf16(const float* __restrict__ x, bf16* __restrict__ y) {
    int i = blockIdx.x * 256 + threadIdx.x;   // i over 512*512
    int kk = i >> 9, n = i & 511;
    y[n * 512 + kk] = (bf16)x[i];
}

// ---------------------------------------------------------------------------
// Generic batched GEMM: C[M,N] = epi(A_f32[M,K] @ B[K,N] + bias)
// B is supplied TRANSPOSED: BwT[N][K] bf16, row stride ldbt -> all staging
// copies are contiguous. Block: 256 threads (8 waves). Tile 128x128, K-step 32.
// Wave grid 4(M) x 2(N); each wave owns 32x64 = 2x4 WMMA accumulators.
// EPI: 0 -> Cf = v; 1 -> Cf = sigmoid(v); 2 -> Cb = (bf16)(v*escale) row-major;
//      3 -> Cb = (bf16)(v) transposed per 2048-row batch: Cb[b][n][k]
// ---------------------------------------------------------------------------
template <int EPI, bool HAS_BIAS>
__global__ __launch_bounds__(256) void gemm_bf16(
    const float* __restrict__ A, const bf16* __restrict__ BwT,
    const float* __restrict__ bias, float* __restrict__ Cf, bf16* __restrict__ Cb,
    int M, int N, int K, int lda, int ldbt, int ldc,
    long long sA, long long sBt, long long sC, float escale,
    int ldct, long long sCt)
{
    __shared__ __align__(16) bf16 ldsA[128][40];   // [m][k] row-major (+pad)
    __shared__ __align__(16) bf16 ldsBT[128][40];  // [n][k] K-contiguous (+pad)

    const int bz = blockIdx.z;
    A   += (long long)bz * sA;
    BwT += (long long)bz * sBt;
    const long long cbase = (long long)bz * sC;

    const int m0 = blockIdx.y * 128, n0 = blockIdx.x * 128;
    const int t = threadIdx.x;
    const int lane = t & 31, wid = t >> 5;
    const int l = lane & 15, hi = lane >> 4;
    const int wm = (wid >> 1) * 32, wn = (wid & 1) * 64;

    f32x8 acc[2][4] = {};

    const int arow = t >> 1, ac0 = (t & 1) * 16;       // A staging: 2 thr/row
    const int bn = t >> 1, bk0 = (t & 1) * 16;         // B staging: 2 thr/N-row
    const float* aptr = A + (long long)(m0 + arow) * lda + ac0;
    const bf16*  bptr = BwT + (long long)(n0 + bn) * ldbt + bk0;

    for (int k0 = 0; k0 < K; k0 += 32) {
        __syncthreads();
        { // stage A: 128x32 f32 -> bf16
            const float* src = aptr + k0;
            bf16* dst = &ldsA[arow][ac0];
            #pragma unroll
            for (int i = 0; i < 16; i += 4) {
                f32x4 v = *(const f32x4*)(src + i);
                dst[i + 0] = (bf16)v.x; dst[i + 1] = (bf16)v.y;
                dst[i + 2] = (bf16)v.z; dst[i + 3] = (bf16)v.w;
            }
        }
        { // stage B from transposed source: contiguous 16-element copy
            const bf16* src = bptr + k0;
            bf16* dst = &ldsBT[bn][bk0];
            *(u32x4*)(dst)     = *(const u32x4*)(src);
            *(u32x4*)(dst + 8) = *(const u32x4*)(src + 8);
        }
        if (k0 + 32 < K) { // warm caches for next tile
            __builtin_prefetch(aptr + k0 + 32, 0, 0);
            __builtin_prefetch(bptr + k0 + 32, 0, 0);
        }
        __syncthreads();

        Frag a0, a1, bb;
        load_a_frag(a0, &ldsA[wm][0], 40, l, hi);
        load_a_frag(a1, &ldsA[wm + 16][0], 40, l, hi);
        #pragma unroll
        for (int j = 0; j < 4; j++) {
            load_b_frag(bb, &ldsBT[wn + j * 16][0], 40, l, hi);
            acc[0][j] = wmma_bf16(a0, bb, acc[0][j]);
            acc[1][j] = wmma_bf16(a1, bb, acc[1][j]);
        }
    }

    // Epilogue. C frag: lane -> N=l, VGPR r -> M = r + 8*hi.
    float bvals[4];
    #pragma unroll
    for (int j = 0; j < 4; j++)
        bvals[j] = HAS_BIAS ? bias[n0 + wn + j * 16 + l] : 0.0f;

    const int rbase = m0 + wm + hi * 8;
    const int cbcol = n0 + wn + l;
    // EPI==3: whole 128-row block lies in one 2048-row batch (2048 % 128 == 0)
    const int tbb = m0 >> 11;
    const long long tbase = (long long)tbb * sCt - (long long)tbb * 2048;
    #pragma unroll
    for (int i = 0; i < 2; i++)
        #pragma unroll
        for (int r = 0; r < 8; r++) {
            const int row = rbase + i * 16 + r;
            const long long rowoff = cbase + (long long)row * ldc + cbcol;
            #pragma unroll
            for (int j = 0; j < 4; j++) {
                const float v = acc[i][j][r] + bvals[j];
                if (EPI == 0)      Cf[rowoff + j * 16] = v;
                else if (EPI == 1) Cf[rowoff + j * 16] = 1.0f / (1.0f + __expf(-v));
                else if (EPI == 2) Cb[rowoff + j * 16] = (bf16)(v * escale);
                else { // transposed store: Cb[b][col][row-in-batch]
                    const long long ti = tbase + (long long)(cbcol + j * 16) * ldct + row;
                    Cb[ti] = (bf16)v;
                }
            }
        }
}

// ---------------------------------------------------------------------------
// Flash attention per (b, head, 64-row q tile). Block = 128 threads = 4 waves,
// each wave owns 16 q rows x full Dh=64. Online softmax over K in 64-key tiles.
// K tile staged via Tensor Data Mover (async DMA + s_wait_tensorcnt); V tile
// staged contiguously from the pre-transposed wvT. Epilogue fuses
// o = g * (attn/rowsum + bias_o).
// ---------------------------------------------------------------------------
__global__ __launch_bounds__(128) void attn_kernel(
    const bf16* __restrict__ wq, const bf16* __restrict__ wk,
    const bf16* __restrict__ wvT, const float* __restrict__ g,
    const float* __restrict__ bias_o, float* __restrict__ o_out)
{
    __shared__ __align__(16) bf16 ldsK[64][72];       // [kidx][d] row-major
    __shared__ __align__(16) bf16 ldsVT[64][72];      // [d][kidx]
    __shared__ __align__(16) bf16 ldsP[4][16][72];    // per-wave P tile

    const int qt = blockIdx.x, h = blockIdx.y, b = blockIdx.z;
    const int t = threadIdx.x, lane = t & 31, wid = t >> 5;
    const int l = lane & 15, hi = lane >> 4;
    const int Q = 2048, KLEN = 2048, HID = 512;
    const long long qrow0 = (long long)b * Q + qt * 64 + wid * 16;
    const long long krow0 = (long long)b * KLEN;

    // wq A fragments (contraction over d=64 -> two 16x32 k-steps), kept in regs
    Frag aq[2];
    {
        const bf16* base = wq + qrow0 * HID + h * 64;
        load_a_frag(aq[0], base, HID, l, hi);
        load_a_frag(aq[1], base + 32, HID, l, hi);
    }

    float mrun[8], lrun[8];
    #pragma unroll
    for (int r = 0; r < 8; r++) { mrun[r] = -1e30f; lrun[r] = 0.0f; }
    f32x8 oacc[4] = {};

#ifndef USE_TDM
    const int krow = t >> 1, kc0 = (t & 1) * 32;   // ldsK staging (fallback)
#endif
    const int vrow = t >> 1, vc0 = (t & 1) * 32;   // ldsVT staging: 2 thr/d-row
    const bf16* kbase = wk + krow0 * HID + h * 64;
    // wvT layout: [b][n=hid][k], row stride KLEN
    const bf16* vbase = wvT + ((long long)b * HID + h * 64 + vrow) * KLEN + vc0;

    for (int k0 = 0; k0 < KLEN; k0 += 64) {
        __syncthreads();
#ifdef USE_TDM
        if (t < 32) { // one wave issues the async K-tile DMA (EXEC-independent)
            tdm_load_2d_bf16((unsigned)(size_t)&ldsK[0][0],
                             kbase + (long long)k0 * HID, 64u, 64u, (unsigned)HID);
        }
#else
        { // stage K tile (row-major copy, 4x b128 per thread)
            const bf16* src = kbase + (long long)(k0 + krow) * HID + kc0;
            bf16* dst = &ldsK[krow][kc0];
            #pragma unroll
            for (int i = 0; i < 32; i += 8)
                *(u32x4*)(dst + i) = *(const u32x4*)(src + i);
        }
#endif
        { // stage V^T tile: contiguous copy from pre-transposed wvT
            const bf16* src = vbase + k0;
            bf16* dst = &ldsVT[vrow][vc0];
            #pragma unroll
            for (int i = 0; i < 32; i += 8)
                *(u32x4*)(dst + i) = *(const u32x4*)(src + i);
        }
        if (k0 + 64 < KLEN) {
            __builtin_prefetch(vbase + k0 + 64, 0, 0);
            __builtin_prefetch(kbase + (long long)(k0 + 64 + (t >> 1)) * HID, 0, 0);
        }
#ifdef USE_TDM
        if (t < 32) __builtin_amdgcn_s_wait_tensorcnt(0);
#endif
        __syncthreads();

        // S(16x64) = wq @ wk^T : B operand rows are kidx, contraction d
        f32x8 sacc[4] = {};
        #pragma unroll
        for (int kt = 0; kt < 4; kt++) {
            Frag bk0f, bk1f;
            load_b_frag(bk0f, &ldsK[kt * 16][0], 72, l, hi);
            load_b_frag(bk1f, &ldsK[kt * 16][32], 72, l, hi);
            sacc[kt] = wmma_bf16(aq[0], bk0f, sacc[kt]);
            sacc[kt] = wmma_bf16(aq[1], bk1f, sacc[kt]);
        }

        // Online softmax: C-frag row = VGPR index -> per-VGPR row stats;
        // row reduction = xor-shuffle over the 16-lane half (masks 1,2,4,8).
        float pv[4][8];
        #pragma unroll
        for (int r = 0; r < 8; r++) {
            float tm = fmaxf(fmaxf(sacc[0][r], sacc[1][r]),
                             fmaxf(sacc[2][r], sacc[3][r]));
            #pragma unroll
            for (int m = 1; m < 16; m <<= 1) tm = fmaxf(tm, __shfl_xor(tm, m, 32));
            const float mnew = fmaxf(mrun[r], tm);
            const float alpha = __expf(mrun[r] - mnew);
            float ts = 0.0f;
            #pragma unroll
            for (int kt = 0; kt < 4; kt++) {
                pv[kt][r] = __expf(sacc[kt][r] - mnew);
                ts += pv[kt][r];
            }
            #pragma unroll
            for (int m = 1; m < 16; m <<= 1) ts += __shfl_xor(ts, m, 32);
            lrun[r] = lrun[r] * alpha + ts;
            mrun[r] = mnew;
            #pragma unroll
            for (int dt = 0; dt < 4; dt++) oacc[dt][r] *= alpha;
        }

        // P: register (C layout) -> LDS -> A fragments (per-wave buffer,
        // same-wave DS ops are kept in order via DScnt)
        #pragma unroll
        for (int kt = 0; kt < 4; kt++)
            #pragma unroll
            for (int r = 0; r < 8; r++)
                ldsP[wid][r + hi * 8][kt * 16 + l] = (bf16)pv[kt][r];

        Frag ap0, ap1;
        load_a_frag(ap0, &ldsP[wid][0][0], 72, l, hi);
        load_a_frag(ap1, &ldsP[wid][0][32], 72, l, hi);
        #pragma unroll
        for (int dt = 0; dt < 4; dt++) {
            Frag bv0, bv1;
            load_b_frag(bv0, &ldsVT[dt * 16][0], 72, l, hi);
            load_b_frag(bv1, &ldsVT[dt * 16][32], 72, l, hi);
            oacc[dt] = wmma_bf16(ap0, bv0, oacc[dt]);
            oacc[dt] = wmma_bf16(ap1, bv1, oacc[dt]);
        }
    }

    // Epilogue: o = g * (attn/l + bias_o)
    const long long ebase = (qrow0 + hi * 8) * HID + h * 64 + l;
    #pragma unroll
    for (int dt = 0; dt < 4; dt++)
        #pragma unroll
        for (int r = 0; r < 8; r++) {
            const long long idx = ebase + (long long)r * HID + dt * 16;
            const float v = oacc[dt][r] / lrun[r] + bias_o[idx];
            o_out[idx] = v * g[idx];
        }
}

// ---------------------------------------------------------------------------
// Host-side orchestration
// ---------------------------------------------------------------------------
extern "C" void kernel_launch(void* const* d_in, const int* in_sizes, int n_in,
                              void* d_out, int out_size, void* d_ws, size_t ws_size,
                              hipStream_t stream) {
    const int B = 4, Q = 2048, KLEN = 2048, D = 512, HID = 512;
    const float SCALE = 0.125f; // 64^-0.5

    const float* q    = (const float*)d_in[0];
    const float* k    = (const float*)d_in[1];
    const float* v    = (const float*)d_in[2];
    const float* bias = (const float*)d_in[3];
    const float* Wq = (const float*)d_in[4];  const float* bq = (const float*)d_in[5];
    const float* Wk = (const float*)d_in[6];  const float* bk = (const float*)d_in[7];
    const float* Wv = (const float*)d_in[8];  const float* bv = (const float*)d_in[9];
    const float* Wg = (const float*)d_in[10]; const float* bg = (const float*)d_in[11];
    const float* Wo = (const float*)d_in[12]; const float* bo = (const float*)d_in[13];
    float* out = (float*)d_out;

    // workspace carve
    size_t off = 0;
    auto carve = [&](size_t bytes) -> void* {
        void* p = (char*)d_ws + off;
        off += (bytes + 255) & ~(size_t)255;
        return p;
    };
    const size_t wsz = (size_t)D * HID;        // 262144
    const size_t msz = (size_t)B * Q * HID;    // 4194304
    bf16* WqT_b = (bf16*)carve(wsz * 2);
    bf16* WkT_b = (bf16*)carve(wsz * 2);
    bf16* WvT_b = (bf16*)carve(wsz * 2);
    bf16* WgT_b = (bf16*)carve(wsz * 2);
    bf16* WoT_b = (bf16*)carve(wsz * 2);
    bf16*  wqp    = (bf16*)carve(msz * 2);   // row-major (B*Q, HID)
    bf16*  wkp    = (bf16*)carve(msz * 2);   // row-major (B*K, HID)
    bf16*  wvT    = (bf16*)carve(msz * 2);   // transposed (B, HID, K)
    float* gbuf   = (float*)carve(msz * 4);
    float* bias_o = (float*)carve(msz * 4);
    float* ogated = (float*)carve(msz * 4);

    // 1) weights -> bf16, transposed (WT[n][k]) for contiguous GEMM staging
    const int cgrid = (int)(wsz / 256);
    cvt_t_bf16<<<cgrid, 256, 0, stream>>>(Wq, WqT_b);
    cvt_t_bf16<<<cgrid, 256, 0, stream>>>(Wk, WkT_b);
    cvt_t_bf16<<<cgrid, 256, 0, stream>>>(Wv, WvT_b);
    cvt_t_bf16<<<cgrid, 256, 0, stream>>>(Wg, WgT_b);
    cvt_t_bf16<<<cgrid, 256, 0, stream>>>(Wo, WoT_b);

    // 2) projections: (B*Q, D) @ (D, HID)
    const int M = B * Q; // 8192
    dim3 pg(HID / 128, M / 128, 1);
    gemm_bf16<2, true><<<pg, 256, 0, stream>>>(q, WqT_b, bq, nullptr, wqp,
        M, HID, D, D, D, HID, 0, 0, 0, SCALE, 0, 0);
    gemm_bf16<2, true><<<pg, 256, 0, stream>>>(k, WkT_b, bk, nullptr, wkp,
        M, HID, D, D, D, HID, 0, 0, 0, 1.0f, 0, 0);
    // V projection: store transposed per batch -> wvT[b][n][k]
    gemm_bf16<3, true><<<pg, 256, 0, stream>>>(v, WvT_b, bv, nullptr, wvT,
        M, HID, D, D, D, HID, 0, 0, 0, 1.0f, KLEN, (long long)HID * KLEN);
    gemm_bf16<1, true><<<pg, 256, 0, stream>>>(q, WgT_b, bg, gbuf, nullptr,
        M, HID, D, D, D, HID, 0, 0, 0, 1.0f, 0, 0);

    // 3) bias_o[b] = bias[b] @ wv[b]  (B operand = wvT, already transposed)
    dim3 bgrid(HID / 128, Q / 128, B);
    gemm_bf16<0, false><<<bgrid, 256, 0, stream>>>(bias, wvT, nullptr, bias_o, nullptr,
        Q, HID, KLEN, KLEN, KLEN, HID,
        (long long)Q * KLEN, (long long)HID * KLEN, (long long)Q * HID, 1.0f, 0, 0);

    // 4) flash attention + fused gate/bias epilogue
    dim3 agrid(Q / 64, 8, B);
    attn_kernel<<<agrid, 128, 0, stream>>>(wqp, wkp, wvT, gbuf, bias_o, ogated);

    // 5) out = ogated @ Wo + bo
    gemm_bf16<0, true><<<pg, 256, 0, stream>>>(ogated, WoT_b, bo, out, nullptr,
        M, D, HID, HID, HID, D, 0, 0, 0, 1.0f, 0, 0);
}
